// Model_74071005987062
// MI455X (gfx1250) — matrix operand order repaired and checked by reading, not compiled, not run on hardware
//
#include <hip/hip_runtime.h>
#include <hip/hip_bf16.h>
#include <math.h>

// Model dims
#define BD 8
#define LD 512
#define DD 128
#define KD 32
#define MD (BD*LD)   // 4096 rows (b,l)
#define ND (DD*KD)   // 4096 cols (d*K+k)
#define CD 256       // concat reduce dim [proxy | x]

typedef __bf16 bf16;
typedef __attribute__((ext_vector_type(16))) __bf16 v16bf;
typedef __attribute__((ext_vector_type(8)))  float  v8f;

// ---------- helpers ----------
__device__ __forceinline__ unsigned short f2bf(float f) {
  unsigned int u = __float_as_uint(f);
  u = (u + 0x7FFFu + ((u >> 16) & 1u)) >> 16;
  return (unsigned short)u;
}
__device__ __forceinline__ float bf2f(unsigned short h) {
  return __uint_as_float((unsigned int)h << 16);
}
__device__ __forceinline__ float fsig(float x) { return 1.0f / (1.0f + __expf(-x)); }
__device__ __forceinline__ float ftanh(float x) {
  x = fminf(fmaxf(x, -15.0f), 15.0f);
  float e = __expf(2.0f * x);
  return (e - 1.0f) / (e + 1.0f);
}

// A fragment (16x32 bf16, M x K): lane holds row m=lane&15.
// elems 0..7  = K[koff + (lane>=16?8:0) .. +7]
// elems 8..15 = K[koff + 16 + (lane>=16?8:0) .. +7]   (ISA 7.12.2)
__device__ __forceinline__ v16bf load_frag_a(const bf16* rowp, int koff, int lane) {
  const bf16* p = rowp + koff + ((lane >> 4) << 3);
  union { v16bf v; uint4 q[2]; } u;
  u.q[0] = *(const uint4*)(p);
  u.q[1] = *(const uint4*)(p + 16);
  return u.v;
}
// B fragment (32x16 bf16, K x N): lane holds column n=lane&15; same K layout as A:
// elems 0..7 = K[koff + (lane>=16?8:0) ..], elems 8..15 = +16. Weight rows (torch (out,in)) are K-contiguous.
__device__ __forceinline__ v16bf load_frag_b(const bf16* rowp, int koff, int lane) {
  const bf16* p = rowp + koff + ((lane >> 4) << 3);
  union { v16bf v; uint4 q[2]; } u;
  u.q[0] = *(const uint4*)(p);
  u.q[1] = *(const uint4*)(p + 16);
  return u.v;
}
__device__ __forceinline__ v8f wmma_bf16(v16bf a, v16bf b, v8f c) {
  return __builtin_amdgcn_wmma_f32_16x16x32_bf16(false, a, false, b, (short)0, c, false, false);
}

// ---------- kernels ----------

// Pack weights to bf16: Bg=[W_g|V_g], Bi=[W_i|V_i] (4096x256), WzB (32*128*128), bias sums.
__global__ void __launch_bounds__(256) k_pack(
    const float* Wg, const float* Vg, const float* Wi, const float* Vi,
    const float* bWg, const float* bVg, const float* bWi, const float* bVi,
    const float* Wz, unsigned short* Bg, unsigned short* Bi,
    unsigned short* WzB, float* biasg, float* biasi) {
  int i = blockIdx.x * 256 + threadIdx.x;
  if (i < ND * CD) {
    int n = i >> 8, c = i & 255;
    float g = (c < DD) ? Wg[n * DD + c] : Vg[n * DD + c - DD];
    float w = (c < DD) ? Wi[n * DD + c] : Vi[n * DD + c - DD];
    Bg[i] = f2bf(g);
    Bi[i] = f2bf(w);
  } else if (i < ND * CD + KD * DD * DD) {
    int j = i - ND * CD;
    WzB[j] = f2bf(Wz[j]);
  } else if (i < ND * CD + KD * DD * DD + ND) {
    int j = i - ND * CD - KD * DD * DD;
    biasg[j] = bWg[j] + bVg[j];
    biasi[j] = bWi[j] + bVi[j];
  }
}

// Abig[m][c] = c<128 ? proxy[m][c] : x[m][c-128]  (bf16, 4096x256)
__global__ void __launch_bounds__(256) k_abig(const float* x, unsigned short* Abig) {
  int i = blockIdx.x * 256 + threadIdx.x;  // M*CD
  int m = i >> 8, c = i & 255;
  float v;
  if (c < DD) {
    int l = m & (LD - 1);
    v = (l == 0) ? 0.0f : x[(size_t)(m - 1) * DD + c];
  } else {
    v = x[(size_t)m * DD + (c - DD)];
  }
  Abig[i] = f2bf(v);
}

// Per-batch: seq_pool -> relu MLP -> omega = max(base + off, 0)
__global__ void __launch_bounds__(128) k_omega(
    const float* x, const float* w1, const float* b1,
    const float* w2, const float* b2, const float* obase, float* omega) {
  __shared__ float pool[DD];
  __shared__ float h[32];
  int b = blockIdx.x, t = threadIdx.x;
  float s = 0.0f;
  for (int l = 0; l < LD; ++l) s += x[((size_t)(b * LD + l)) * DD + t];
  pool[t] = s * (1.0f / LD);
  __syncthreads();
  if (t < 32) {
    float a = b1[t];
    for (int c = 0; c < DD; ++c) a += w1[t * DD + c] * pool[c];
    h[t] = fmaxf(a, 0.0f);
  }
  __syncthreads();
  if (t < KD) {
    float a = b2[t];
    for (int c = 0; c < 32; ++c) a += w2[t * 32 + c] * h[c];
    omega[b * KD + t] = fmaxf(obase[t] + a, 0.0f);
  }
}

// cos/sin tables (B,L,K)
__global__ void __launch_bounds__(256) k_trig(const float* omega, float* cosT, float* sinT) {
  int i = blockIdx.x * 256 + threadIdx.x;  // B*L*K
  int k = i & 31, l = (i >> 5) & (LD - 1), b = i >> 14;
  float t = (float)(l + 1) * (1.0f / LD);
  float ph = omega[b * KD + k] * t;
  float s, c;
  __sincosf(ph, &s, &c);
  cosT[i] = c;
  sinT[i] = s;
}

// Per-row small GEMMs: f_ste(K), f_fre(D), o_pre(K)=v_term+w_term+biases, attn(K softmax)
__global__ void __launch_bounds__(256) k_rows(
    const float* x,
    const float* Wste, const float* bWste, const float* Vste, const float* bVste,
    const float* Wfre, const float* bWfre, const float* Vfre, const float* bVfre,
    const float* Vo, const float* Wo, const float* bWo, const float* bo,
    const float* faw, const float* fab,
    float* f_ste, float* f_fre, float* o_pre, float* attn) {
  __shared__ float xr[DD], pr[DD], lg[KD];
  int m = blockIdx.x, t = threadIdx.x;
  int l = m & (LD - 1);
  if (t < DD) {
    xr[t] = x[(size_t)m * DD + t];
    pr[t] = (l == 0) ? 0.0f : x[(size_t)(m - 1) * DD + t];
  }
  __syncthreads();
  if (t < KD) {
    float a = bWste[t] + bVste[t];
    for (int c = 0; c < DD; ++c) a += Wste[t * DD + c] * pr[c] + Vste[t * DD + c] * xr[c];
    f_ste[(size_t)m * KD + t] = fsig(a);
  } else if (t < KD + DD) {
    int j = t - KD;
    float a = bWfre[j] + bVfre[j];
    for (int c = 0; c < DD; ++c) a += Wfre[j * DD + c] * pr[c] + Vfre[j * DD + c] * xr[c];
    f_fre[(size_t)m * DD + j] = fsig(a);
  } else if (t < KD + DD + KD) {
    int j = t - KD - DD;
    float a = bWo[j] + bo[j];
    for (int c = 0; c < DD; ++c) a += Wo[j * DD + c] * pr[c] + Vo[j * DD + c] * xr[c];
    o_pre[(size_t)m * KD + j] = a;
  } else if (t < KD + DD + KD + KD) {
    int j = t - KD - DD - KD;
    float a = fab[j];
    for (int c = 0; c < DD; ++c) a += faw[j * DD + c] * xr[c];
    lg[j] = a;
  }
  __syncthreads();
  if (t < KD) {
    float mx = -1e30f;
    for (int j = 0; j < KD; ++j) mx = fmaxf(mx, lg[j]);
    float sm = 0.0f;
    for (int j = 0; j < KD; ++j) sm += __expf(lg[j] - mx);
    attn[(size_t)m * KD + t] = __expf(lg[t] - mx) / sm;
  }
}

// WMMA GEMM: gi[m][n] = sigmoid(Abig@Bg^T + bias_g) * tanh(Abig@Bi^T + bias_i)
// wave = 16 rows x 64 cols, dual accumulators; block = 8 waves (128 rows). grid (64, 32).
__global__ void __launch_bounds__(256) k_gemm_gi(
    const unsigned short* Abig_, const unsigned short* Bg_, const unsigned short* Bi_,
    const float* biasg, const float* biasi, float* gi) {
  const bf16* A  = (const bf16*)Abig_;
  const bf16* BG = (const bf16*)Bg_;
  const bf16* BI = (const bf16*)Bi_;
  int lane = threadIdx.x & 31, wid = threadIdx.x >> 5;
  int m0 = blockIdx.y * 128 + wid * 16;
  int n0 = blockIdx.x * 64;
  const bf16* arow = A + (size_t)(m0 + (lane & 15)) * CD;
  v8f zz = {0.f, 0.f, 0.f, 0.f, 0.f, 0.f, 0.f, 0.f};
  v8f accg[4] = {zz, zz, zz, zz};
  v8f acci[4] = {zz, zz, zz, zz};
#pragma unroll
  for (int kt = 0; kt < CD; kt += 32) {
    v16bf a = load_frag_a(arow, kt, lane);
#pragma unroll
    for (int t = 0; t < 4; ++t) {
      const bf16* bgrow = BG + (size_t)(n0 + 16 * t + (lane & 15)) * CD;
      accg[t] = wmma_bf16(a, load_frag_b(bgrow, kt, lane), accg[t]);
      const bf16* birow = BI + (size_t)(n0 + 16 * t + (lane & 15)) * CD;
      acci[t] = wmma_bf16(a, load_frag_b(birow, kt, lane), acci[t]);
    }
  }
  int hi = (lane >> 4) << 3;
#pragma unroll
  for (int t = 0; t < 4; ++t) {
    int col = n0 + 16 * t + (lane & 15);
    float bg = biasg[col], bi = biasi[col];
#pragma unroll
    for (int r = 0; r < 8; ++r) {
      int row = m0 + r + hi;
      float g = fsig(accg[t][r] + bg);
      float iv = ftanh(acci[t][r] + bi);
      gi[(size_t)row * ND + col] = g * iv;
    }
  }
}

// Linear recurrence scan along L; lanes over k (coalesced gi reads).
// 8 timesteps buffered in LDS, then transposed write-out so A lands as
// As[b,l,k,d] bf16 (d-contiguous for the z-GEMM A fragments), one uint4
// store per thread per chunk. global_prefetch on the next gi chunk hides
// the sequential-path latency.
// block = 256 threads = (8 d) x (32 k); grid = B * (D/8) = 128.
__global__ void __launch_bounds__(256) k_scan(
    const float* gi, const float* f_fre, const float* f_ste,
    const float* cosT, const float* sinT, unsigned short* As) {
  __shared__ float tile[8][KD][9];  // stride 9 -> conflict-free
  int t = threadIdx.x;
  int b = blockIdx.x >> 4;
  int d0 = (blockIdx.x & 15) << 3;
  int dl = t >> 5;       // 0..7
  int d = d0 + dl;
  int k = t & 31;
  int lbw = t >> 5, k2 = t & 31;  // write-out mapping: (timestep, k-row)
  float re = 0.0f, im = 0.0f;
  for (int l0 = 0; l0 < LD; l0 += 8) {
#pragma unroll
    for (int lb = 0; lb < 8; ++lb) {
      int m = b * LD + l0 + lb;
      if (l0 + 8 < LD)  // prefetch next chunk's gi element (full cacheline)
        __builtin_prefetch(&gi[(size_t)(m + 8) * ND + d * KD + k], 0, 1);
      float f = f_fre[(size_t)m * DD + d] * f_ste[(size_t)m * KD + k];
      float a = gi[(size_t)m * ND + d * KD + k];
      float c = cosT[(size_t)m * KD + k];
      float s = sinT[(size_t)m * KD + k];
      re = f * re + a * c;
      im = f * im + a * s;
      tile[lb][k][dl] = sqrtf(re * re + im * im + 1e-8f);
    }
    __syncthreads();
    {
      int m = b * LD + l0 + lbw;
      union { uint4 q; unsigned short s[8]; } pk;
#pragma unroll
      for (int j = 0; j < 8; ++j) pk.s[j] = f2bf(tile[lbw][k2][j]);
      *(uint4*)(&As[((size_t)m * KD + k2) * DD + d0]) = pk.q;
    }
    __syncthreads();
  }
}

// u_amp + o + attn*o -> wbuf (B,L,K); vectorized bf16 reads.
__global__ void __launch_bounds__(256) k_uamp(
    const unsigned short* As, const float* Uo,
    const float* o_pre, const float* attn, float* wbuf) {
  int i = blockIdx.x * 256 + threadIdx.x;  // M*K
  int m = i >> 5, k = i & 31;
  const unsigned short* a = As + ((size_t)m * KD + k) * DD;
  const float* u = Uo + k * DD;
  float s = 0.0f;
#pragma unroll
  for (int d8 = 0; d8 < DD; d8 += 8) {
    union { uint4 q; unsigned short h[8]; } v;
    v.q = *(const uint4*)(a + d8);
#pragma unroll
    for (int j = 0; j < 8; ++j) s += bf2f(v.h[j]) * u[d8 + j];
  }
  float o = fsig(s + o_pre[i]);
  wbuf[i] = attn[i] * o;
}

// Batched-over-k WMMA z GEMM with fused tanh + weighted accumulation into out.
// wave = 16 rows x 32 cols; block = 8 waves (128 rows); grid (4, 32).
__global__ void __launch_bounds__(256) k_zout(
    const unsigned short* As_, const unsigned short* WzB_, const float* bz,
    const float* wbuf, float* out) {
  const bf16* A = (const bf16*)As_;
  const bf16* W = (const bf16*)WzB_;
  int lane = threadIdx.x & 31, wid = threadIdx.x >> 5;
  int m0 = blockIdx.y * 128 + wid * 16;
  int n0 = blockIdx.x * 32;
  int hi = (lane >> 4) << 3;
  v8f zz = {0.f, 0.f, 0.f, 0.f, 0.f, 0.f, 0.f, 0.f};
  v8f out0 = zz, out1 = zz;
  for (int k = 0; k < KD; ++k) {
    v8f z0 = zz, z1 = zz;
    const bf16* arow  = A + ((size_t)(m0 + (lane & 15)) * KD + k) * DD;
    const bf16* brow0 = W + ((size_t)(k * DD + n0 + (lane & 15))) * DD;
    const bf16* brow1 = W + ((size_t)(k * DD + n0 + 16 + (lane & 15))) * DD;
#pragma unroll
    for (int kt = 0; kt < DD; kt += 32) {
      v16bf a = load_frag_a(arow, kt, lane);
      z0 = wmma_bf16(a, load_frag_b(brow0, kt, lane), z0);
      z1 = wmma_bf16(a, load_frag_b(brow1, kt, lane), z1);
    }
    float b0 = bz[k * DD + n0 + (lane & 15)];
    float b1 = bz[k * DD + n0 + 16 + (lane & 15)];
#pragma unroll
    for (int r = 0; r < 8; ++r) {
      float wv = wbuf[(size_t)(m0 + r + hi) * KD + k];
      out0[r] += wv * ftanh(z0[r] + b0);
      out1[r] += wv * ftanh(z1[r] + b1);
    }
  }
#pragma unroll
  for (int r = 0; r < 8; ++r) {
    int row = m0 + r + hi;
    out[(size_t)row * DD + n0 + (lane & 15)]      = out0[r];
    out[(size_t)row * DD + n0 + 16 + (lane & 15)] = out1[r];
  }
}

// ---------- launch ----------
extern "C" void kernel_launch(void* const* d_in, const int* in_sizes, int n_in,
                              void* d_out, int out_size, void* d_ws, size_t ws_size,
                              hipStream_t stream) {
  const float* x     = (const float*)d_in[0];
  const float* obase = (const float*)d_in[1];
  const float* oa_w1 = (const float*)d_in[2];
  const float* oa_b1 = (const float*)d_in[3];
  const float* oa_w2 = (const float*)d_in[4];
  const float* oa_b2 = (const float*)d_in[5];
  const float* Wste  = (const float*)d_in[6];
  const float* bWste = (const float*)d_in[7];
  const float* Vste  = (const float*)d_in[8];
  const float* bVste = (const float*)d_in[9];
  const float* Wfre  = (const float*)d_in[10];
  const float* bWfre = (const float*)d_in[11];
  const float* Vfre  = (const float*)d_in[12];
  const float* bVfre = (const float*)d_in[13];
  const float* Wg    = (const float*)d_in[14];
  const float* bWg   = (const float*)d_in[15];
  const float* Vg    = (const float*)d_in[16];
  const float* bVg   = (const float*)d_in[17];
  const float* Wi    = (const float*)d_in[18];
  const float* bWi   = (const float*)d_in[19];
  const float* Vi    = (const float*)d_in[20];
  const float* bVi   = (const float*)d_in[21];
  const float* Uo    = (const float*)d_in[22];
  const float* Vo    = (const float*)d_in[23];
  const float* Wo    = (const float*)d_in[24];
  const float* bWo   = (const float*)d_in[25];
  const float* bo    = (const float*)d_in[26];
  const float* Wz    = (const float*)d_in[27];
  const float* bz    = (const float*)d_in[28];
  const float* faw   = (const float*)d_in[29];
  const float* fab   = (const float*)d_in[30];

  char* p = (char*)d_ws;
  auto alloc = [&](size_t bytes) -> char* {
    char* r = p;
    p += (bytes + 255) & ~(size_t)255;
    return r;
  };
  float* omega = (float*)alloc(BD * KD * 4);
  float* cosT  = (float*)alloc((size_t)BD * LD * KD * 4);
  float* sinT  = (float*)alloc((size_t)BD * LD * KD * 4);
  float* f_ste = (float*)alloc((size_t)MD * KD * 4);
  float* f_fre = (float*)alloc((size_t)MD * DD * 4);
  float* o_pre = (float*)alloc((size_t)MD * KD * 4);
  float* attn  = (float*)alloc((size_t)MD * KD * 4);
  float* wbuf  = (float*)alloc((size_t)MD * KD * 4);
  unsigned short* Abig = (unsigned short*)alloc((size_t)MD * CD * 2);
  unsigned short* Bg   = (unsigned short*)alloc((size_t)ND * CD * 2);
  unsigned short* Bi   = (unsigned short*)alloc((size_t)ND * CD * 2);
  float* biasg = (float*)alloc(ND * 4);
  float* biasi = (float*)alloc(ND * 4);
  unsigned short* WzB  = (unsigned short*)alloc((size_t)KD * DD * DD * 2);
  float* gi = (float*)alloc((size_t)MD * ND * 4);           // 64 MB
  unsigned short* As = (unsigned short*)alloc((size_t)MD * KD * DD * 2);  // 32 MB
  float* out = (float*)d_out;

  k_pack<<<(ND * CD + KD * DD * DD + ND) / 256, 256, 0, stream>>>(
      Wg, Vg, Wi, Vi, bWg, bVg, bWi, bVi, Wz, Bg, Bi, WzB, biasg, biasi);
  k_abig<<<(MD * CD) / 256, 256, 0, stream>>>(x, Abig);
  k_omega<<<BD, 128, 0, stream>>>(x, oa_w1, oa_b1, oa_w2, oa_b2, obase, omega);
  k_trig<<<(BD * LD * KD) / 256, 256, 0, stream>>>(omega, cosT, sinT);
  k_rows<<<MD, 256, 0, stream>>>(x, Wste, bWste, Vste, bVste, Wfre, bWfre, Vfre, bVfre,
                                 Vo, Wo, bWo, bo, faw, fab, f_ste, f_fre, o_pre, attn);
  k_gemm_gi<<<dim3(ND / 64, MD / 128), 256, 0, stream>>>(Abig, Bg, Bi, biasg, biasi, gi);
  k_scan<<<BD * (DD / 8), 256, 0, stream>>>(gi, f_fre, f_ste, cosT, sinT, As);
  k_uamp<<<(MD * KD) / 256, 256, 0, stream>>>(As, Uo, o_pre, attn, wbuf);
  k_zout<<<dim3(DD / 32, MD / 128), 256, 0, stream>>>(As, WzB, bz, wbuf, out);
}
